// SimpleRNN_4939212390684
// MI455X (gfx1250) — matrix operand order repaired and checked by reading, not compiled, not run on hardware
//
#include <hip/hip_runtime.h>
#include <math.h>

typedef __bf16 bf16_t;
typedef __attribute__((ext_vector_type(16))) __bf16 v16bf;
typedef __attribute__((ext_vector_type(8)))  __bf16 v8bf;
typedef __attribute__((ext_vector_type(8)))  float  v8f;

#define Hh    128
#define Dk    100
#define Bsz   256
#define Tlen  512
#define LDST  136   // LDS row stride (bf16 elems): 272B/row -> 68 dwords -> bank offset 4/row

// ---------------- weight conversion: fp32 -> bf16 (pad W_ih0 K: 100 -> 128) --------------
__global__ void prep_weights(const float* __restrict__ Wih0, const float* __restrict__ Whh0,
                             const float* __restrict__ Wih1, const float* __restrict__ Whh1,
                             bf16_t* __restrict__ wsb) {
    int i = blockIdx.x * blockDim.x + threadIdx.x;   // 0 .. 4*128*128-1
    int mat = i >> 14;
    int idx = i & 16383;
    int r = idx >> 7, c = idx & 127;
    float f;
    if      (mat == 0) f = (c < Dk) ? Wih0[r * Dk + c] : 0.0f;
    else if (mat == 1) f = Whh0[idx];
    else if (mat == 2) f = Wih1[idx];
    else               f = Whh1[idx];
    wsb[i] = (bf16_t)f;
}

// ---------------- layer0: fused embedding gather + input projection (WMMA GEMM) ---------
// xp[t, b, h] = sum_d emb[x[b,t], d] * W_ih0[h, d] + b_ih0[h] + b_hh0[h]
__global__ void embed_proj0(const int* __restrict__ x, const float* __restrict__ emb,
                            const bf16_t* __restrict__ Wb, const float* __restrict__ b_ih,
                            const float* __restrict__ b_hh, float* __restrict__ xp) {
    const int lane = threadIdx.x & 31, wid = threadIdx.x >> 5;
    const int m0 = (blockIdx.x * 8 + wid) * 16;          // flat (b*T + t) row base
    const int col = lane & 15, hs = lane >> 4;
    const int bb = m0 >> 9;                              // batch index (T=512 divides tile)
    const int t0 = m0 & 511;                             // time base; rows span t0..t0+15
    const int token = x[m0 + col];                       // A row r = lane&15

    // A fragments (16x32 bf16, contiguous k = 16*hs + v), zero-padded K beyond D=100
    v16bf a[4];
    #pragma unroll
    for (int kk = 0; kk < 4; ++kk) {
        #pragma unroll
        for (int v = 0; v < 16; ++v) {
            int k = kk * 32 + 16 * hs + v;
            float f = (k < Dk) ? emb[(size_t)token * Dk + k] : 0.0f;
            a[kk][v] = (bf16_t)f;
        }
    }
    #pragma unroll
    for (int nt = 0; nt < 8; ++nt) {
        const int n0 = nt * 16;
        const float bias = b_ih[n0 + col] + b_hh[n0 + col];
        v8f acc;
        #pragma unroll
        for (int i = 0; i < 8; ++i) acc[i] = bias;
        #pragma unroll
        for (int kk = 0; kk < 4; ++kk) {
            // B fragment: B[k][n] = W[n][k]; 16 contiguous K per half-wave (32B load)
            v16bf b = *(const v16bf*)(Wb + (size_t)(n0 + col) * Hh + kk * 32 + 16 * hs);
            acc = __builtin_amdgcn_wmma_f32_16x16x32_bf16(false, a[kk], false, b,
                                                          (short)0, acc, false, false);
        }
        #pragma unroll
        for (int i = 0; i < 8; ++i) {
            const int t = t0 + i + 8 * hs;
            xp[((size_t)t * Bsz + bb) * Hh + n0 + col] = acc[i];   // [T,B,H] scatter
        }
    }
}

// ---------------- layer1 input projection from bf16 y1 (WMMA GEMM) ----------------------
__global__ void proj1(const bf16_t* __restrict__ y1, const bf16_t* __restrict__ Wb,
                      const float* __restrict__ b_ih, const float* __restrict__ b_hh,
                      float* __restrict__ xp) {
    const int lane = threadIdx.x & 31, wid = threadIdx.x >> 5;
    const int m0 = (blockIdx.x * 8 + wid) * 16;
    const int col = lane & 15, hs = lane >> 4;
    const int bb = m0 >> 9;
    const int t0 = m0 & 511;
    const bf16_t* arow = y1 + (size_t)(m0 + col) * Hh;

    v16bf a[4];
    #pragma unroll
    for (int kk = 0; kk < 4; ++kk) {
        v8bf r0 = *(const v8bf*)(arow + kk * 32 + 16 * hs);       // K = +16*hs+0..7 (half-wave)
        v8bf r1 = *(const v8bf*)(arow + kk * 32 + 16 * hs + 8);   // K = +16*hs+8..15
        #pragma unroll
        for (int v = 0; v < 8; ++v) { a[kk][v] = r0[v]; a[kk][v + 8] = r1[v]; }
    }
    #pragma unroll
    for (int nt = 0; nt < 8; ++nt) {
        const int n0 = nt * 16;
        const float bias = b_ih[n0 + col] + b_hh[n0 + col];
        v8f acc;
        #pragma unroll
        for (int i = 0; i < 8; ++i) acc[i] = bias;
        #pragma unroll
        for (int kk = 0; kk < 4; ++kk) {
            v16bf b = *(const v16bf*)(Wb + (size_t)(n0 + col) * Hh + kk * 32 + 16 * hs);
            acc = __builtin_amdgcn_wmma_f32_16x16x32_bf16(false, a[kk], false, b,
                                                          (short)0, acc, false, false);
        }
        #pragma unroll
        for (int i = 0; i < 8; ++i) {
            const int t = t0 + i + 8 * hs;
            xp[((size_t)t * Bsz + bb) * Hh + n0 + col] = acc[i];
        }
    }
}

// ---------------- sequential scan: h = tanh(xp[t] + h @ W_hh^T) --------------------------
// One block per 16-batch-row tile; 8 waves each own 16 hidden cols; W_hh fragments stay
// resident in VGPRs for all T steps; h double-buffered in LDS as bf16; xp (in [T,B,H])
// is software-pipelined one step ahead into registers so its L2 latency hides behind
// the barrier + LDS exchange instead of sitting on the serial critical path.
template<bool STORE_Y, bool STORE_H>
__global__ void rnn_scan(const float* __restrict__ xp, const bf16_t* __restrict__ Whh,
                         bf16_t* __restrict__ y_out, float* __restrict__ h_out) {
    __shared__ bf16_t hbuf[2][16 * LDST];
    const int lane = threadIdx.x & 31, wid = threadIdx.x >> 5;
    const int col = lane & 15, hs = lane >> 4;
    const int n0 = wid * 16;
    const int b0 = blockIdx.x * 16;

    // resident B fragments: B[k][n] = W_hh[n][k]
    v16bf bw[4];
    #pragma unroll
    for (int kk = 0; kk < 4; ++kk)
        bw[kk] = *(const v16bf*)(Whh + (size_t)(n0 + col) * Hh + kk * 32 + 16 * hs);

    for (int i = threadIdx.x; i < 16 * LDST; i += blockDim.x) {
        hbuf[0][i] = (bf16_t)0.0f;
        hbuf[1][i] = (bf16_t)0.0f;
    }

    // prime the xp pipeline: step 0 tile is one contiguous 8KB slab at t=0
    const float* xpl = xp + ((size_t)0 * Bsz + b0) * Hh + n0 + col;
    v8f xnext;
    #pragma unroll
    for (int i = 0; i < 8; ++i) xnext[i] = xpl[(size_t)(i + 8 * hs) * Hh];

    __syncthreads();

    for (int t = 0; t < Tlen; ++t) {
        const int cur = t & 1, nxt = cur ^ 1;
        v8f acc = xnext;                       // C init = precomputed input projection
        // h @ W_hh^T via 4 WMMAs, A streamed from LDS
        #pragma unroll
        for (int kk = 0; kk < 4; ++kk) {
            const bf16_t* ap = &hbuf[cur][(lane & 15) * LDST + kk * 32 + 16 * hs];
            v8bf r0 = *(const v8bf*)ap;
            v8bf r1 = *(const v8bf*)(ap + 8);
            v16bf a;
            #pragma unroll
            for (int v = 0; v < 8; ++v) { a[v] = r0[v]; a[v + 8] = r1[v]; }
            acc = __builtin_amdgcn_wmma_f32_16x16x32_bf16(false, a, false, bw[kk],
                                                          (short)0, acc, false, false);
        }
        // prefetch next step's xp tile (off the dependency chain; waits land next iter)
        if (t + 1 < Tlen) {
            const float* xn = xp + ((size_t)(t + 1) * Bsz + b0) * Hh + n0 + col;
            #pragma unroll
            for (int i = 0; i < 8; ++i) xnext[i] = xn[(size_t)(i + 8 * hs) * Hh];
        }
        #pragma unroll
        for (int i = 0; i < 8; ++i) {
            const float hv = tanhf(acc[i]);
            const int m = i + 8 * hs;
            const bf16_t hb = (bf16_t)hv;
            hbuf[nxt][m * LDST + n0 + col] = hb;
            if (STORE_Y)
                y_out[((size_t)(b0 + m) * Tlen + t) * Hh + n0 + col] = hb;
            if (STORE_H && t == Tlen - 1)
                h_out[(size_t)(b0 + m) * Hh + n0 + col] = hv;
        }
        __syncthreads();
    }
}

// ---------------- final FC: [256,128] x [128,2] ------------------------------------------
__global__ void fc_out(const float* __restrict__ h2, const float* __restrict__ fw,
                       const float* __restrict__ fb, float* __restrict__ out) {
    int b = blockIdx.x * blockDim.x + threadIdx.x;
    if (b >= Bsz) return;
    float s0 = fb[0], s1 = fb[1];
    #pragma unroll 8
    for (int k = 0; k < Hh; ++k) {
        float h = h2[(size_t)b * Hh + k];
        s0 += h * fw[k];
        s1 += h * fw[Hh + k];
    }
    out[b * 2 + 0] = s0;
    out[b * 2 + 1] = s1;
}

extern "C" void kernel_launch(void* const* d_in, const int* in_sizes, int n_in,
                              void* d_out, int out_size, void* d_ws, size_t ws_size,
                              hipStream_t stream) {
    const int*   x     = (const int*)  d_in[0];
    const float* emb   = (const float*)d_in[1];
    const float* Wih0  = (const float*)d_in[2];
    const float* Whh0  = (const float*)d_in[3];
    const float* bih0  = (const float*)d_in[4];
    const float* bhh0  = (const float*)d_in[5];
    const float* Wih1  = (const float*)d_in[6];
    const float* Whh1  = (const float*)d_in[7];
    const float* bih1  = (const float*)d_in[8];
    const float* bhh1  = (const float*)d_in[9];
    const float* fcw   = (const float*)d_in[10];
    const float* fcb   = (const float*)d_in[11];
    float* out = (float*)d_out;

    // workspace layout
    char* ws = (char*)d_ws;
    bf16_t* wsb   = (bf16_t*)ws;                 // 4 bf16 weight mats, 128x128 each
    bf16_t* Wih0b = wsb + 0 * 16384;
    bf16_t* Whh0b = wsb + 1 * 16384;
    bf16_t* Wih1b = wsb + 2 * 16384;
    bf16_t* Whh1b = wsb + 3 * 16384;
    size_t off = 4 * 16384 * sizeof(bf16_t);     // 131072 B
    float*  xp = (float*)(ws + off);             // [T, B, H] f32, reused for both layers
    off += (size_t)Bsz * Tlen * Hh * sizeof(float);
    bf16_t* y1 = (bf16_t*)(ws + off);            // [B*T, H] bf16
    off += (size_t)Bsz * Tlen * Hh * sizeof(bf16_t);
    float*  h2 = (float*)(ws + off);             // [B, H] f32

    const int mtiles = (Bsz * Tlen) / 16;        // 8192 -> 1024 blocks of 8 waves

    prep_weights<<<(4 * 16384) / 256, 256, 0, stream>>>(Wih0, Whh0, Wih1, Whh1, wsb);
    embed_proj0<<<mtiles / 8, 256, 0, stream>>>(x, emb, Wih0b, bih0, bhh0, xp);
    rnn_scan<true, false><<<Bsz / 16, 256, 0, stream>>>(xp, Whh0b, y1, nullptr);
    proj1<<<mtiles / 8, 256, 0, stream>>>(y1, Wih1b, bih1, bhh1, xp);
    rnn_scan<false, true><<<Bsz / 16, 256, 0, stream>>>(xp, Whh1b, nullptr, h2);
    fc_out<<<1, Bsz, 0, stream>>>(h2, fcw, fcb, out);
}
